// VQEmbeddingEMA_30829275251373
// MI455X (gfx1250) — matrix operand order repaired and checked by reading, not compiled, not run on hardware
//
#include <hip/hip_runtime.h>
#include <hip/hip_bf16.h>
#include <stdint.h>

// ---------------- problem constants ----------------
#define M_CODES 512
#define D_EMB   128
#define N_TOK   65536          // 32 * 2048
#define TILE_TOK 128
#define DECAY   0.999f
#define OMD     0.001f         // 1 - DECAY
#define EPSV    1e-5f
#define COMMIT  0.25f

// ---------------- workspace layout (float offsets) ----------------
#define WS_COUNTS 0            // 512
#define WS_DW     512          // 512*128 = 65536
#define WS_LOSS   66048        // 1
#define WS_MSUM   66049        // 1
#define WS_EN     66050        // 512  (||e||^2)
#define WS_EBF    66564        // bf16[512*128] = 131072 bytes (16B aligned)
#define WS_ZERO_N 66050        // floats to zero each launch

// ---------------- output layout (float offsets) ----------------
#define OUT_QST  0             // 32*2048*128 = 8388608
#define OUT_LOSS 8388608       // 1
#define OUT_EMB  8388609       // 65536
#define OUT_CNT  8454145       // 512
#define OUT_WGT  8454657       // 65536

typedef __attribute__((ext_vector_type(16))) __bf16 v16bf;
typedef __attribute__((ext_vector_type(8)))  float  v8f;

union BPack { uint4 q[2]; v16bf v; };
union BF2U  { __bf16 h; unsigned short u; };

// ============================================================
// Kernel 0: zero the accumulated scratch (deterministic per call)
// ============================================================
__global__ void vq_init(float* __restrict__ ws) {
  int i = blockIdx.x * 256 + threadIdx.x;
  if (i < WS_ZERO_N) ws[i] = 0.0f;
}

// ============================================================
// Kernel 1: prep — ||e||^2, bf16 codebook copy, sum(mask)
// blocks 0..511: one embedding row each; blocks 512..575: mask sum
// ============================================================
__global__ void vq_prep(const float* __restrict__ emb,
                        const float* __restrict__ x_mask,
                        float* __restrict__ ws) {
  __shared__ float red[128];
  const int tid = threadIdx.x;
  if (blockIdx.x < M_CODES) {
    const int row = blockIdx.x;
    float e = emb[row * D_EMB + tid];
    BF2U b; b.h = (__bf16)e;
    ((unsigned short*)(ws + WS_EBF))[row * D_EMB + tid] = b.u;
    red[tid] = e * e;
    __syncthreads();
    #pragma unroll
    for (int s = 64; s > 0; s >>= 1) {
      if (tid < s) red[tid] += red[tid + s];
      __syncthreads();
    }
    if (tid == 0) ws[WS_EN + row] = red[0];
  } else {
    const int mb = blockIdx.x - M_CODES;      // 0..63, each covers 1024 mask vals
    const float* mp = x_mask + mb * 1024 + tid * 8;
    float s = 0.0f;
    #pragma unroll
    for (int j = 0; j < 8; ++j) s += mp[j];
    red[tid] = s;
    __syncthreads();
    #pragma unroll
    for (int st = 64; st > 0; st >>= 1) {
      if (tid < st) red[tid] += red[tid + st];
      __syncthreads();
    }
    if (tid == 0) atomicAdd(&ws[WS_MSUM], red[0]);
  }
}

// ============================================================
// Kernel 2: main — WMMA distance + argmin + gather + scatter
// grid = 512 blocks, 256 threads (8 wave32), 128 tokens/block
// ============================================================
#define EROW_PITCH 272   // 128 bf16 (256B) + 16B pad -> conflict-friendly, 16B aligned

__launch_bounds__(256)
__global__ void vq_main(const float* __restrict__ x,
                        const float* __restrict__ x_mask,
                        const float* __restrict__ emb,
                        float* __restrict__ ws,
                        float* __restrict__ out) {
  extern __shared__ __align__(16) char smem[];
  char*  sh_e   = smem;                                  // 512 * 272 B
  float* sh_en  = (float*)(smem + M_CODES * EROW_PITCH); // 512 f32
  int*   sh_idx = (int*)(smem + M_CODES * EROW_PITCH + 2048); // 128 ints

  const int tid  = threadIdx.x;
  const int lane = tid & 31;
  const int wave = tid >> 5;

  // stage ||e||^2 (plain DS path)
  sh_en[tid]       = ws[WS_EN + tid];
  sh_en[tid + 256] = ws[WS_EN + tid + 256];

  // stage bf16 codebook via CDNA5 async global->LDS DMA
  {
    const char* ebf = (const char*)(ws + WS_EBF);
    #pragma unroll 4
    for (int it = 0; it < 32; ++it) {
      int q   = it * 256 + tid;                 // 16B chunk id, 8192 total
      int row = q >> 4, c16 = q & 15;
      unsigned int lds_off = (unsigned int)(size_t)(sh_e + row * EROW_PITCH + c16 * 16);
      const char* gsrc = ebf + q * 16;
      asm volatile("global_load_async_to_lds_b128 %0, %1, off"
                   :: "v"(lds_off), "v"(gsrc) : "memory");
    }
    asm volatile("s_wait_asynccnt 0" ::: "memory");
  }
  __syncthreads();

  // ---- build A fragments: 16 tokens per wave, K=128 in 4 chunks of 32 ----
  // 16-bit A layout: lane m (0-15) holds row m, K = {0..7,16..23}+32c;
  //                  lane m+16 holds row m, K = {8..15,24..31}+32c.
  const int m    = lane & 15;
  const int gtok = blockIdx.x * TILE_TOK + wave * 16 + m;
  const float mv = x_mask[gtok];                // [B,1,T] flat == token order
  v16bf aF[4];
  {
    const float* xr = x + (size_t)gtok * D_EMB + ((lane < 16) ? 0 : 8);
    #pragma unroll
    for (int c = 0; c < 4; ++c) {
      const float* p = xr + c * 32;
      float4 p0 = *(const float4*)(p);
      float4 p1 = *(const float4*)(p + 4);
      float4 p2 = *(const float4*)(p + 16);
      float4 p3 = *(const float4*)(p + 20);
      aF[c][0]  = (__bf16)(p0.x * mv); aF[c][1]  = (__bf16)(p0.y * mv);
      aF[c][2]  = (__bf16)(p0.z * mv); aF[c][3]  = (__bf16)(p0.w * mv);
      aF[c][4]  = (__bf16)(p1.x * mv); aF[c][5]  = (__bf16)(p1.y * mv);
      aF[c][6]  = (__bf16)(p1.z * mv); aF[c][7]  = (__bf16)(p1.w * mv);
      aF[c][8]  = (__bf16)(p2.x * mv); aF[c][9]  = (__bf16)(p2.y * mv);
      aF[c][10] = (__bf16)(p2.z * mv); aF[c][11] = (__bf16)(p2.w * mv);
      aF[c][12] = (__bf16)(p3.x * mv); aF[c][13] = (__bf16)(p3.y * mv);
      aF[c][14] = (__bf16)(p3.z * mv); aF[c][15] = (__bf16)(p3.w * mv);
    }
  }

  // ---- sweep 32 code tiles: software-pipelined B loads + 4 WMMAs each ----
  float minv[8];
  int   mini[8];
  #pragma unroll
  for (int r = 0; r < 8; ++r) { minv[r] = 3.4e38f; mini[r] = 0; }

  const int koffB = (lane < 16) ? 0 : 16;      // B frag K-half byte offset

  uint4 bq[2][8];                              // ping-pong B fragment buffers
  {
    const char* rowp = sh_e + m * EROW_PITCH + koffB;   // tile 0
    #pragma unroll
    for (int c = 0; c < 4; ++c) {
      bq[0][2 * c]     = *(const uint4*)(rowp + c * 64);
      bq[0][2 * c + 1] = *(const uint4*)(rowp + c * 64 + 32);
    }
  }

  #pragma unroll 2
  for (int nt = 0; nt < 32; ++nt) {
    const int n = nt * 16 + m;                 // this lane's code column
    uint4* cur = bq[nt & 1];
    uint4* nxt = bq[(nt + 1) & 1];
    // prefetch next tile's B fragments while this tile's WMMAs run
    if (nt < 31) {
      const char* rowp = sh_e + (n + 16) * EROW_PITCH + koffB;
      #pragma unroll
      for (int c = 0; c < 4; ++c) {
        nxt[2 * c]     = *(const uint4*)(rowp + c * 64);
        nxt[2 * c + 1] = *(const uint4*)(rowp + c * 64 + 32);
      }
    }
    const float en_n = sh_en[n];

    v8f acc = {0.f, 0.f, 0.f, 0.f, 0.f, 0.f, 0.f, 0.f};
    #pragma unroll
    for (int c = 0; c < 4; ++c) {
      BPack bp;
      bp.q[0] = cur[2 * c];
      bp.q[1] = cur[2 * c + 1];
      acc = __builtin_amdgcn_wmma_f32_16x16x32_bf16(
                false, aF[c], false, bp.v, (short)0, acc, false, false);
    }
    #pragma unroll
    for (int r = 0; r < 8; ++r) {              // C layout: lanes<16 rows 0-7, else 8-15
      float d = en_n - 2.0f * acc[r];
      if (d < minv[r]) { minv[r] = d; mini[r] = n; }
    }
  }

  // ---- cross-lane argmin within each 16-lane half (ties -> lower index) ----
  #pragma unroll
  for (int s = 1; s <= 8; s <<= 1) {
    #pragma unroll
    for (int r = 0; r < 8; ++r) {
      float ov = __shfl_xor(minv[r], s, 32);
      int   oi = __shfl_xor(mini[r], s, 32);
      if (ov < minv[r] || (ov == minv[r] && oi < mini[r])) { minv[r] = ov; mini[r] = oi; }
    }
  }
  #pragma unroll
  for (int r = 0; r < 8; ++r) {
    if ((lane & 15) == r) {
      int mm = (lane < 16) ? r : (8 + r);
      sh_idx[wave * 16 + mm] = mini[r];
    }
  }
  __syncthreads();

  // ---- gather + q_st + dw/count scatter + loss partial ----
  {
    const int tk   = tid >> 1;                 // 0..127
    const int half = tid & 1;                  // 64 features each
    const int idx  = sh_idx[tk];
    const int g    = blockIdx.x * TILE_TOK + tk;
    const float mvt = x_mask[g];
    const float* erow = emb + (size_t)idx * D_EMB + half * 64;
    const float* xrow = x   + (size_t)g   * D_EMB + half * 64;
    float* qrow  = out + OUT_QST + (size_t)g   * D_EMB + half * 64;
    float* dwrow = ws  + WS_DW   + (size_t)idx * D_EMB + half * 64;

    if (half == 0) atomicAdd(&ws[WS_COUNTS + idx], 1.0f);

    float lacc = 0.0f;
    #pragma unroll
    for (int d = 0; d < 64; d += 4) {
      float4 e4 = *(const float4*)(erow + d);
      float4 x4 = *(const float4*)(xrow + d);
      float4 q4;
      q4.x = e4.x * mvt; q4.y = e4.y * mvt; q4.z = e4.z * mvt; q4.w = e4.w * mvt;
      *(float4*)(qrow + d) = q4;               // q_st = quantized * mask
      float xm0 = x4.x * mvt, xm1 = x4.y * mvt, xm2 = x4.z * mvt, xm3 = x4.w * mvt;
      atomicAdd(dwrow + d + 0, xm0);
      atomicAdd(dwrow + d + 1, xm1);
      atomicAdd(dwrow + d + 2, xm2);
      atomicAdd(dwrow + d + 3, xm3);
      float d0 = xm0 * mvt - q4.x, d1 = xm1 * mvt - q4.y;
      float d2 = xm2 * mvt - q4.z, d3 = xm3 * mvt - q4.w;
      lacc += d0 * d0 + d1 * d1 + d2 * d2 + d3 * d3;
    }
    #pragma unroll
    for (int s = 16; s > 0; s >>= 1) lacc += __shfl_down(lacc, s, 32);
    if (lane == 0) atomicAdd(&ws[WS_LOSS], lacc);
  }
}

// ============================================================
// Kernel 3: finalize — EMA update, smoothing, loss scale
// ============================================================
__launch_bounds__(512)
__global__ void vq_finalize(const float* __restrict__ ema_count,
                            const float* __restrict__ ema_weight,
                            float* __restrict__ ws,
                            float* __restrict__ out) {
  __shared__ float red[M_CODES];
  const int mm = threadIdx.x;
  float c = DECAY * ema_count[mm] + OMD * ws[WS_COUNTS + mm];
  red[mm] = c;
  __syncthreads();
  #pragma unroll
  for (int s = 256; s > 0; s >>= 1) {
    if (mm < s) red[mm] += red[mm + s];
    __syncthreads();
  }
  const float n  = red[0];
  const float nc = (c + EPSV) / (n + (float)M_CODES * EPSV) * n;
  out[OUT_CNT + mm] = nc;
  const float inv = 1.0f / nc;
  #pragma unroll
  for (int d = 0; d < D_EMB; d += 4) {
    float4 w4  = *(const float4*)(ema_weight + mm * D_EMB + d);
    float4 dw4 = *(const float4*)(ws + WS_DW + mm * D_EMB + d);
    float4 nw;
    nw.x = DECAY * w4.x + OMD * dw4.x; nw.y = DECAY * w4.y + OMD * dw4.y;
    nw.z = DECAY * w4.z + OMD * dw4.z; nw.w = DECAY * w4.w + OMD * dw4.w;
    *(float4*)(out + OUT_WGT + mm * D_EMB + d) = nw;
    float4 ne;
    ne.x = nw.x * inv; ne.y = nw.y * inv; ne.z = nw.z * inv; ne.w = nw.w * inv;
    *(float4*)(out + OUT_EMB + mm * D_EMB + d) = ne;
  }
  if (mm == 0)
    out[OUT_LOSS] = COMMIT * ws[WS_LOSS] / (ws[WS_MSUM] * (float)D_EMB);
}

// ============================================================
extern "C" void kernel_launch(void* const* d_in, const int* in_sizes, int n_in,
                              void* d_out, int out_size, void* d_ws, size_t ws_size,
                              hipStream_t stream) {
  const float* x          = (const float*)d_in[0];
  const float* x_mask     = (const float*)d_in[1];
  const float* emb        = (const float*)d_in[2];
  const float* ema_count  = (const float*)d_in[3];
  const float* ema_weight = (const float*)d_in[4];
  float* out = (float*)d_out;
  float* ws  = (float*)d_ws;

  vq_init<<<(WS_ZERO_N + 255) / 256, 256, 0, stream>>>(ws);
  vq_prep<<<M_CODES + 64, 128, 0, stream>>>(emb, x_mask, ws);
  const size_t shmem = (size_t)M_CODES * EROW_PITCH + 2048 + TILE_TOK * sizeof(int);
  vq_main<<<N_TOK / TILE_TOK, 256, shmem, stream>>>(x, x_mask, emb, ws, out);
  vq_finalize<<<1, M_CODES, 0, stream>>>(ema_count, ema_weight, ws, out);
}
